// TemporalGraphConvolution_20014547599384
// MI455X (gfx1250) — matrix operand order repaired and checked by reading, not compiled, not run on hardware
//
#include <hip/hip_runtime.h>
#include <hip/hip_bf16.h>
#include <math.h>

typedef __attribute__((ext_vector_type(4)))  __bf16 v4bf;
typedef __attribute__((ext_vector_type(8)))  __bf16 v8bf;
typedef __attribute__((ext_vector_type(16))) __bf16 v16bf;
typedef __attribute__((ext_vector_type(8)))  float  v8f;

#define D128 128
#define EPSLN 1e-5f

// ---------------------------------------------------------------------------
// Small helpers
// ---------------------------------------------------------------------------
__device__ __forceinline__ float wave_sum32(float v) {
#pragma unroll
  for (int off = 16; off > 0; off >>= 1) v += __shfl_xor(v, off, 32);
  return v;
}
__device__ __forceinline__ float sigmf(float x) { return 1.0f / (1.0f + __expf(-x)); }

// ---------------------------------------------------------------------------
// Converters
// ---------------------------------------------------------------------------
__global__ void cvt_f32_bf16(const float* __restrict__ src, __bf16* __restrict__ dst, int n) {
  int i = blockIdx.x * blockDim.x + threadIdx.x;
  if (i < n) dst[i] = (__bf16)src[i];
}

// gcn_weight is [K=128, M=128]; we need B as [M, K] row-major (NT GEMM form)
__global__ void transpose_cvt_128(const float* __restrict__ src, __bf16* __restrict__ dst) {
  int m = blockIdx.x;      // 0..127
  int k = threadIdx.x;     // 0..127
  dst[m * D128 + k] = (__bf16)src[k * D128 + m];
}

// skip_W is [128, 256]; split into Wh=[:, :128] and Wx=[:, 128:], both [M=128,K=128]
__global__ void cvt_skip_split(const float* __restrict__ skw,
                               __bf16* __restrict__ wh, __bf16* __restrict__ wx) {
  int m = blockIdx.x;
  int k = threadIdx.x;
  wh[m * D128 + k] = (__bf16)skw[m * 256 + k];
  wx[m * D128 + k] = (__bf16)skw[m * 256 + 128 + k];
}

__global__ void zero_f32(float* __restrict__ p, int n) {
  int i = blockIdx.x * blockDim.x + threadIdx.x;
  if (i < n) p[i] = 0.0f;
}

// ---------------------------------------------------------------------------
// NT GEMM: C[R,M](f32) = A[R,K](bf16, row-major) @ B[M,K](bf16, row-major)^T
// Block = 128 threads (4 wave32), tile = 64 rows x 64 cols, K step = 32.
// Uses v_wmma_f32_16x16x32_bf16 with LDS-staged tiles; fragments are built
// with ds_load_b128 per the CDNA5 wave32 VGPR layouts.
// ---------------------------------------------------------------------------
__global__ __launch_bounds__(128)
void gemm_nt_bf16(const __bf16* __restrict__ A, const __bf16* __restrict__ B,
                  float* __restrict__ C, int R, int K, int M, int accum) {
  __shared__ __bf16 aSh[64 * 32];
  __shared__ __bf16 bSh[64 * 32];

  const int t    = threadIdx.x;       // 0..127
  const int wave = t >> 5;            // 0..3 -> row sub-tile
  const int lane = t & 31;
  const int n    = lane & 15;
  const int h    = lane >> 4;         // lane half
  const int rowBase = blockIdx.x * 64;
  const int colBase = blockIdx.y * 64;

  v8f acc[4] = {v8f(0.f), v8f(0.f), v8f(0.f), v8f(0.f)};

  union Frag { v16bf v; v8bf p[2]; };

  const int ksteps = K >> 5;
  for (int ks = 0; ks < ksteps; ++ks) {
    const int k0 = ks << 5;
    // Cooperative tile load: 64x32 bf16 = 256 chunks of 16B; 2 A + 2 B per thread.
#pragma unroll
    for (int it = 0; it < 2; ++it) {
      int c    = t + it * 128;        // 0..255
      int r    = c >> 2;              // tile row 0..63
      int koff = (c & 3) << 3;        // 0,8,16,24
      v8bf av = v8bf(__bf16(0.f));
      int grow = rowBase + r;
      if (grow < R)
        av = *reinterpret_cast<const v8bf*>(A + (size_t)grow * K + k0 + koff);
      *reinterpret_cast<v8bf*>(&aSh[r * 32 + koff]) = av;
      v8bf bv = *reinterpret_cast<const v8bf*>(B + (size_t)(colBase + r) * K + k0 + koff);
      *reinterpret_cast<v8bf*>(&bSh[r * 32 + koff]) = bv;
    }
    if (ks + 1 < ksteps) {
      // speculative prefetch of next B K-slab -> global_prefetch_b8
      __builtin_prefetch(B + (size_t)(colBase + (t >> 1)) * K + k0 + 32, 0, 1);
    }
    __syncthreads();

    // A fragment (16x32): elems 0-7 <- K=8h.., elems 8-15 <- K=16+8h..
    Frag af;
    const int am = wave * 16 + n;
    af.p[0] = *reinterpret_cast<const v8bf*>(&aSh[am * 32 + h * 8]);
    af.p[1] = *reinterpret_cast<const v8bf*>(&aSh[am * 32 + 16 + h * 8]);

#pragma unroll
    for (int c4 = 0; c4 < 4; ++c4) {
      // B fragment (32x16): elems 0-7 <- K=8h.., elems 8-15 <- K=16+8h.. at column bn
      Frag bf;
      const int bn = c4 * 16 + n;
      bf.p[0] = *reinterpret_cast<const v8bf*>(&bSh[bn * 32 + h * 8]);
      bf.p[1] = *reinterpret_cast<const v8bf*>(&bSh[bn * 32 + 16 + h * 8]);
      acc[c4] = __builtin_amdgcn_wmma_f32_16x16x32_bf16(
          false, af.v, false, bf.v, (short)0, acc[c4], false, false);
    }
    __syncthreads();
  }

  // C/D layout: VGPR r -> M=r+8h, lane&15 -> N
#pragma unroll
  for (int c4 = 0; c4 < 4; ++c4) {
#pragma unroll
    for (int r = 0; r < 8; ++r) {
      int grow = rowBase + wave * 16 + h * 8 + r;
      if (grow < R) {
        size_t idx = (size_t)grow * M + colBase + c4 * 16 + n;
        float v = acc[c4][r];
        if (accum) v += C[idx];
        C[idx] = v;
      }
    }
  }
}

// ---------------------------------------------------------------------------
// SpMM scatter: agg[row] += val * support[col]  (one wave per edge)
// ---------------------------------------------------------------------------
__global__ __launch_bounds__(256)
void spmm_scatter(const int* __restrict__ erow, const int* __restrict__ ecol,
                  const float* __restrict__ eval, const float* __restrict__ support,
                  float* __restrict__ agg, int E) {
  int e = blockIdx.x * 8 + (threadIdx.x >> 5);
  if (e >= E) return;
  int lane = threadIdx.x & 31;
  int r = erow[e];
  int c = ecol[e];
  float v = eval[e];
  const float* s = support + (size_t)c * D128;
  float* a = agg + (size_t)r * D128;
#pragma unroll
  for (int j = 0; j < 4; ++j) {
    float d = v * s[lane + 32 * j];
    __hip_atomic_fetch_add(&a[lane + 32 * j], d, __ATOMIC_RELAXED, __HIP_MEMORY_SCOPE_AGENT);
  }
}

// ---------------------------------------------------------------------------
// xg = LayerNorm(relu(agg + gcn_bias)) * g + b   -> bf16   (wave per row)
// ---------------------------------------------------------------------------
__global__ __launch_bounds__(256)
void gcn_post_ln(const float* __restrict__ agg, const float* __restrict__ bias,
                 const float* __restrict__ g, const float* __restrict__ b,
                 __bf16* __restrict__ xgb, int Nn) {
  int row = blockIdx.x * 8 + (threadIdx.x >> 5);
  if (row >= Nn) return;
  int lane = threadIdx.x & 31;
  int c = lane * 4;
  float4 a4 = *reinterpret_cast<const float4*>(agg + (size_t)row * D128 + c);
  float4 b4 = *reinterpret_cast<const float4*>(bias + c);
  float v0 = fmaxf(a4.x + b4.x, 0.f);
  float v1 = fmaxf(a4.y + b4.y, 0.f);
  float v2 = fmaxf(a4.z + b4.z, 0.f);
  float v3 = fmaxf(a4.w + b4.w, 0.f);
  float sum = wave_sum32(v0 + v1 + v2 + v3);
  float sq  = wave_sum32(v0 * v0 + v1 * v1 + v2 * v2 + v3 * v3);
  float mean = sum * (1.0f / 128.0f);
  float var  = sq * (1.0f / 128.0f) - mean * mean;
  float rstd = rsqrtf(var + EPSLN);
  float4 g4 = *reinterpret_cast<const float4*>(g + c);
  float4 bb = *reinterpret_cast<const float4*>(b + c);
  v4bf o;
  o[0] = (__bf16)((v0 - mean) * rstd * g4.x + bb.x);
  o[1] = (__bf16)((v1 - mean) * rstd * g4.y + bb.y);
  o[2] = (__bf16)((v2 - mean) * rstd * g4.z + bb.z);
  o[3] = (__bf16)((v3 - mean) * rstd * g4.w + bb.w);
  *reinterpret_cast<v4bf*>(xgb + (size_t)row * D128 + c) = o;
}

// ---------------------------------------------------------------------------
// GRU cell (gate order r,z,n) + LayerNorm -> h bf16   (wave per row)
// ---------------------------------------------------------------------------
__global__ __launch_bounds__(256)
void gru_ln(const float* __restrict__ gi, const float* __restrict__ gh,
            const float* __restrict__ bih, const float* __restrict__ bhh,
            const float* __restrict__ h0,
            const float* __restrict__ g, const float* __restrict__ b,
            __bf16* __restrict__ hb, int Nn) {
  int row = blockIdx.x * 8 + (threadIdx.x >> 5);
  if (row >= Nn) return;
  int lane = threadIdx.x & 31;
  int c = lane * 4;
  const float* giR = gi + (size_t)row * 384;
  const float* ghR = gh + (size_t)row * 384;
  float4 ir  = *reinterpret_cast<const float4*>(giR + c);
  float4 iz  = *reinterpret_cast<const float4*>(giR + 128 + c);
  float4 inn = *reinterpret_cast<const float4*>(giR + 256 + c);
  float4 hr  = *reinterpret_cast<const float4*>(ghR + c);
  float4 hz  = *reinterpret_cast<const float4*>(ghR + 128 + c);
  float4 hn  = *reinterpret_cast<const float4*>(ghR + 256 + c);
  float4 bir = *reinterpret_cast<const float4*>(bih + c);
  float4 biz = *reinterpret_cast<const float4*>(bih + 128 + c);
  float4 bin = *reinterpret_cast<const float4*>(bih + 256 + c);
  float4 bhr = *reinterpret_cast<const float4*>(bhh + c);
  float4 bhz = *reinterpret_cast<const float4*>(bhh + 128 + c);
  float4 bhn = *reinterpret_cast<const float4*>(bhh + 256 + c);
  float4 h4  = *reinterpret_cast<const float4*>(h0 + (size_t)row * D128 + c);
  float hv[4];
#define GRU_COMP(i, X)                                                       \
  {                                                                          \
    float rr = sigmf(ir.X + bir.X + hr.X + bhr.X);                           \
    float zz = sigmf(iz.X + biz.X + hz.X + bhz.X);                           \
    float nn = tanhf(inn.X + bin.X + rr * (hn.X + bhn.X));                   \
    hv[i] = (1.0f - zz) * nn + zz * h4.X;                                    \
  }
  GRU_COMP(0, x) GRU_COMP(1, y) GRU_COMP(2, z) GRU_COMP(3, w)
#undef GRU_COMP
  float sum = wave_sum32(hv[0] + hv[1] + hv[2] + hv[3]);
  float sq  = wave_sum32(hv[0]*hv[0] + hv[1]*hv[1] + hv[2]*hv[2] + hv[3]*hv[3]);
  float mean = sum * (1.0f / 128.0f);
  float var  = sq * (1.0f / 128.0f) - mean * mean;
  float rstd = rsqrtf(var + EPSLN);
  float4 g4 = *reinterpret_cast<const float4*>(g + c);
  float4 bb = *reinterpret_cast<const float4*>(b + c);
  v4bf o;
  o[0] = (__bf16)((hv[0] - mean) * rstd * g4.x + bb.x);
  o[1] = (__bf16)((hv[1] - mean) * rstd * g4.y + bb.y);
  o[2] = (__bf16)((hv[2] - mean) * rstd * g4.z + bb.z);
  o[3] = (__bf16)((hv[3] - mean) * rstd * g4.w + bb.w);
  *reinterpret_cast<v4bf*>(hb + (size_t)row * D128 + c) = o;
}

// ---------------------------------------------------------------------------
// y = leaky_relu(y + skip_b)   (in place on d_out)
// ---------------------------------------------------------------------------
__global__ void bias_leaky(float* __restrict__ y, const float* __restrict__ bias, int total) {
  int i = blockIdx.x * blockDim.x + threadIdx.x;
  if (i < total) {
    float v = y[i] + bias[i & 127];
    y[i] = v >= 0.0f ? v : 0.01f * v;
  }
}

// ---------------------------------------------------------------------------
// Host launcher
// ---------------------------------------------------------------------------
extern "C" void kernel_launch(void* const* d_in, const int* in_sizes, int n_in,
                              void* d_out, int out_size, void* d_ws, size_t ws_size,
                              hipStream_t stream) {
  const float* x    = (const float*)d_in[0];
  const int*   erow = (const int*)d_in[1];
  const int*   ecol = (const int*)d_in[2];
  const float* ev   = (const float*)d_in[3];
  const float* gcnW = (const float*)d_in[4];
  const float* gcnB = (const float*)d_in[5];
  const float* ln1g = (const float*)d_in[6];
  const float* ln1b = (const float*)d_in[7];
  const float* Wih  = (const float*)d_in[8];
  const float* Whh  = (const float*)d_in[9];
  const float* bih  = (const float*)d_in[10];
  const float* bhh  = (const float*)d_in[11];
  const float* ln2g = (const float*)d_in[12];
  const float* ln2b = (const float*)d_in[13];
  const float* skW  = (const float*)d_in[14];
  const float* skb  = (const float*)d_in[15];
  const float* h0   = (const float*)d_in[16];

  const int N = in_sizes[0] / D128;
  const int E = in_sizes[1];

  char* base = (char*)d_ws;
  size_t off = 0;
  auto carve = [&](size_t bytes) -> char* {
    char* p = base + off;
    off += (bytes + 255) & ~(size_t)255;
    return p;
  };

  __bf16* xb    = (__bf16*)carve((size_t)N * D128 * 2);
  __bf16* h0b   = (__bf16*)carve((size_t)N * D128 * 2);
  __bf16* xgb   = (__bf16*)carve((size_t)N * D128 * 2);
  __bf16* hb    = (__bf16*)carve((size_t)N * D128 * 2);
  __bf16* gcnWT = (__bf16*)carve(128 * 128 * 2);
  __bf16* WihB  = (__bf16*)carve(384 * 128 * 2);
  __bf16* WhhB  = (__bf16*)carve(384 * 128 * 2);
  __bf16* sWh   = (__bf16*)carve(128 * 128 * 2);
  __bf16* sWx   = (__bf16*)carve(128 * 128 * 2);
  // Overlapped big region: phase 1 = support+agg (2 x N*128 f32),
  //                        phase 2 = gi+gh (2 x N*384 f32). Union-sized.
  char* big = carve((size_t)N * 384 * 4 * 2);
  float* support = (float*)big;
  float* agg     = (float*)(big + (size_t)N * 384 * 4);
  float* gi      = (float*)big;
  float* gh      = (float*)(big + (size_t)N * 384 * 4);
  float* y = (float*)d_out;

  const int nd = N * D128;

  // --- precision conversion + weight prep ---
  cvt_f32_bf16<<<(nd + 255) / 256, 256, 0, stream>>>(x, xb, nd);
  cvt_f32_bf16<<<(nd + 255) / 256, 256, 0, stream>>>(h0, h0b, nd);
  cvt_f32_bf16<<<(384 * 128 + 255) / 256, 256, 0, stream>>>(Wih, WihB, 384 * 128);
  cvt_f32_bf16<<<(384 * 128 + 255) / 256, 256, 0, stream>>>(Whh, WhhB, 384 * 128);
  transpose_cvt_128<<<128, 128, 0, stream>>>(gcnW, gcnWT);
  cvt_skip_split<<<128, 128, 0, stream>>>(skW, sWh, sWx);
  zero_f32<<<(nd + 255) / 256, 256, 0, stream>>>(agg, nd);

  dim3 gemmBlk(128);
  dim3 g128((N + 63) / 64, 2);   // M = 128
  dim3 g384((N + 63) / 64, 6);   // M = 384

  // --- GCN: support = x @ W ---
  gemm_nt_bf16<<<g128, gemmBlk, 0, stream>>>(xb, gcnWT, support, N, 128, 128, 0);
  // --- SpMM aggregation ---
  spmm_scatter<<<(E + 7) / 8, 256, 0, stream>>>(erow, ecol, ev, support, agg, E);
  // --- relu + bias + LN1 -> xg (bf16) ---
  gcn_post_ln<<<(N + 7) / 8, 256, 0, stream>>>(agg, gcnB, ln1g, ln1b, xgb, N);
  // --- GRU input/hidden GEMMs ---
  gemm_nt_bf16<<<g384, gemmBlk, 0, stream>>>(xgb, WihB, gi, N, 128, 384, 0);
  gemm_nt_bf16<<<g384, gemmBlk, 0, stream>>>(h0b, WhhB, gh, N, 128, 384, 0);
  // --- GRU gates + LN2 -> h (bf16) ---
  gru_ln<<<(N + 7) / 8, 256, 0, stream>>>(gi, gh, bih, bhh, h0, ln2g, ln2b, hb, N);
  // --- skip: y = h @ Wh^T + xg @ Wx^T (accumulate into d_out) ---
  gemm_nt_bf16<<<g128, gemmBlk, 0, stream>>>(hb, sWh, y, N, 128, 128, 0);
  gemm_nt_bf16<<<g128, gemmBlk, 0, stream>>>(xgb, sWx, y, N, 128, 128, 1);
  // --- bias + leaky relu ---
  bias_leaky<<<(nd + 255) / 256, 256, 0, stream>>>(y, skb, nd);
}
